// Attention_36361193128703
// MI455X (gfx1250) — matrix operand order repaired and checked by reading, not compiled, not run on hardware
//
#include <hip/hip_runtime.h>

#define BB 2
#define LL 2048
#define DD 1024
#define HH 16
#define HD 64
#define KK 16
#define NQKV (BB * HH * LL * HD)   // 4,194,304 bf16 elements per tensor
#define NBIAS (HH * (2 * LL - 1))  // 65,520 floats
#define NX (BB * LL * DD)          // 4,194,304 x elements
#define NW (3 * DD * DD)           // 3,145,728 w elements
#define AROWH 36                   // LDS row stride in halves: 32 data + 4 pad (18 dw)

typedef __attribute__((ext_vector_type(16))) __bf16 v16bf;
typedef __attribute__((ext_vector_type(8)))  float  v8f;
typedef __attribute__((ext_vector_type(4)))  unsigned int v4u;
typedef __attribute__((ext_vector_type(8)))  int    v8i;
typedef __attribute__((ext_vector_type(4)))  int    v4i;

__device__ __forceinline__ unsigned short f2bf(float f) {
    __bf16 b = (__bf16)f;
    union { __bf16 b; unsigned short u; } cv;
    cv.b = b;
    return cv.u;
}

__device__ __forceinline__ float half16_max(float v) {
#pragma unroll
    for (int off = 1; off < 16; off <<= 1) v = fmaxf(v, __shfl_xor(v, off, 32));
    return v;
}
__device__ __forceinline__ float half16_sum(float v) {
#pragma unroll
    for (int off = 1; off < 16; off <<= 1) v += __shfl_xor(v, off, 32);
    return v;
}

// ---------------------------------------------------------------------------
// TDM: async 2D bf16 tile load Global -> LDS (D# per ISA 8.3/8.4).
// data_size=2B (code 1); rows of 32 halves (64B) padded with 2 dwords:
// pad_interval=16 dwords (code 3), pad_amount=2 dwords (code 1)
// -> LDS row stride = 18 dwords (bank-conflict-free fragment reads, 8B aligned)
// ---------------------------------------------------------------------------
__device__ __forceinline__ void tdm_load_2d_bf16(unsigned lds_off,
                                                 const unsigned short* gptr,
                                                 unsigned tile_w,   // elems (halves)
                                                 unsigned tile_h,   // rows
                                                 unsigned stride) { // row stride, elems
    unsigned long long ga = (unsigned long long)(const void*)gptr;
    v4u g0;
    g0[0] = 1u;                                   // count=1, user descriptor
    g0[1] = lds_off;                              // LDS byte address
    g0[2] = (unsigned)(ga & 0xFFFFFFFFu);         // global_addr[31:0]
    g0[3] = (unsigned)((ga >> 32) & 0x01FFFFFFu)  // global_addr[56:32]
            | 0x80000000u;                        // type=2 ("image")
    v8i g1;
    // data_size=2B (1), pad_enable, pad_interval=16 dwords (code 3),
    // pad_amount=2 dwords (code 1), workgroup_mask=0
    g1[0] = (int)((1u << 16) | (1u << 20) | (3u << 22) | (1u << 25));
    g1[1] = (int)((tile_w & 0xFFFFu) << 16);      // tensor_dim0[15:0]
    g1[2] = (int)(((tile_h & 0xFFFFu) << 16) | ((tile_w >> 16) & 0xFFFFu));
    g1[3] = (int)(((tile_w & 0xFFFFu) << 16) | ((tile_h >> 16) & 0xFFFFu)); // tile_dim0
    g1[4] = (int)(tile_h & 0xFFFFu);              // tile_dim1 (tile_dim2=0)
    g1[5] = (int)stride;                          // tensor_dim0_stride[31:0]
    g1[6] = 0;
    g1[7] = 0;
    v4i g2 = (v4i){0, 0, 0, 0};
    v4i g3 = (v4i){0, 0, 0, 0};
    v8i g4 = (v8i){0, 0, 0, 0, 0, 0, 0, 0};
    __builtin_amdgcn_tensor_load_to_lds(g0, g1, g2, g3, g4, 0);
}

// ---------------------------------------------------------------------------
// K-1: one-shot f32 -> bf16 conversion pass (bandwidth-bound, ~2us)
// ---------------------------------------------------------------------------
__global__ void cvt_bf16_kernel(const float* __restrict__ src,
                                unsigned short* __restrict__ dst, int n4) {
    int i = blockIdx.x * blockDim.x + threadIdx.x;
    if (i >= n4) return;
    float4 f = ((const float4*)src)[i];
    union { unsigned short us[4]; uint2 d; } r;
    r.us[0] = f2bf(f.x);
    r.us[1] = f2bf(f.y);
    r.us[2] = f2bf(f.z);
    r.us[3] = f2bf(f.w);
    ((uint2*)dst)[i] = r.d;
}

// ---------------------------------------------------------------------------
// K0: precompute TISA bias table: bias[h][r], r in [0, 2L-1), rel = r-(L-1)
// ---------------------------------------------------------------------------
__global__ void tisa_bias_kernel(const float* __restrict__ off,
                                 const float* __restrict__ amp,
                                 const float* __restrict__ shp,
                                 float* __restrict__ bias) {
    int idx = blockIdx.x * blockDim.x + threadIdx.x;
    if (idx >= NBIAS) return;
    int h = idx / (2 * LL - 1);
    int r = idx % (2 * LL - 1);
    float rel = (float)(r - (LL - 1));
    float s = 0.0f;
#pragma unroll
    for (int j = 0; j < KK; ++j) {
        float d = off[h * KK + j] - rel;
        s += amp[h * KK + j] * __expf(-fabsf(shp[h * KK + j]) * d * d);
    }
    bias[idx] = s;
}

// ---------------------------------------------------------------------------
// K1: projection GEMM  proj = x @ w^T  (TDM-staged bf16, pure ds_load + WMMA)
// Workgroup tile 128(M) x 64(N): 8 waves, each 16x64.  Double-buffered TDM.
// scatter: cols [0,D)->k [B,H,L,HD], [D,2D)->v^T [B,H,HD,L], [2D,3D)->q scaled
// ---------------------------------------------------------------------------
__global__ __launch_bounds__(256) void proj_kernel(
        const unsigned short* __restrict__ xb,
        const unsigned short* __restrict__ wb,
        unsigned short* __restrict__ qbuf,
        unsigned short* __restrict__ kbuf,
        unsigned short* __restrict__ vtbuf) {
    __shared__ unsigned short ldsA[2][128 * AROWH];  // 18,432 B
    __shared__ unsigned short ldsB[2][64 * AROWH];   //  9,216 B

    const int lane = threadIdx.x & 31;
    const int lo = lane & 15;
    const int hi = lane >> 4;
    const int widx = threadIdx.x >> 5;
    const int bm = blockIdx.x / 48;         // 32 M blocks of 128
    const int bn = blockIdx.x % 48;         // 48 N blocks of 64
    const int row0 = bm * 128;
    const int col0 = bn * 64;

    v8f c[4];
#pragma unroll
    for (int t = 0; t < 4; ++t) c[t] = (v8f){0.f,0.f,0.f,0.f,0.f,0.f,0.f,0.f};

    // prologue: fetch k-slice 0 into buffer 0
    if (widx == 0) {
        tdm_load_2d_bf16((unsigned)(size_t)&ldsA[0][0], xb + (size_t)row0 * DD, 32, 128, DD);
        tdm_load_2d_bf16((unsigned)(size_t)&ldsB[0][0], wb + (size_t)col0 * DD, 32, 64, DD);
    }

    for (int i = 0; i < 32; ++i) {
        const int buf = i & 1;
        if (widx == 0) {
            if (i < 31) {   // prefetch next k-slice into other buffer
                const int k1 = (i + 1) * 32;
                tdm_load_2d_bf16((unsigned)(size_t)&ldsA[buf ^ 1][0],
                                 xb + (size_t)row0 * DD + k1, 32, 128, DD);
                tdm_load_2d_bf16((unsigned)(size_t)&ldsB[buf ^ 1][0],
                                 wb + (size_t)col0 * DD + k1, 32, 64, DD);
                __builtin_amdgcn_s_wait_tensorcnt(2);  // current slice complete
            } else {
                __builtin_amdgcn_s_wait_tensorcnt(0);
            }
        }
        __syncthreads();

        // A fragment: rows widx*16 + lo; VGPR pair p -> k (bf16 pairs, no cvt)
        const unsigned short* As = &ldsA[buf][(widx * 16 + lo) * AROWH];
        union { v16bf v; unsigned u[8]; } a;
#pragma unroll
        for (int p = 0; p < 8; ++p) {
            int kk = 16 * hi + 2 * p;
            a.u[p] = *(const unsigned*)(As + kk);
        }
        // All 4 B fragments up front -> one DS batch, then 4 back-to-back WMMAs
        union { v16bf v; uint2 d2[4]; } b[4];
#pragma unroll
        for (int t = 0; t < 4; ++t) {
            const unsigned short* Bs = &ldsB[buf][(t * 16 + lo) * AROWH + 16 * hi];
#pragma unroll
            for (int e = 0; e < 4; ++e) b[t].d2[e] = *(const uint2*)(Bs + 4 * e);
        }
#pragma unroll
        for (int t = 0; t < 4; ++t) {
            c[t] = __builtin_amdgcn_wmma_f32_16x16x32_bf16(
                false, a.v, false, b[t].v, (short)0, c[t], false, false);
        }
        __syncthreads();
    }

    // scatter D-matrix: value at (m = r + 8*hi, n = lo)
    const float qscale = 0.125f;  // 1/sqrt(HD)
#pragma unroll
    for (int t = 0; t < 4; ++t) {
        int e = col0 + t * 16 + lo;       // global proj column
        int sect = e >> 10;               // 0=k, 1=v, 2=q
        int cc = e & 1023;
        int h = cc >> 6;
        int d = cc & 63;
#pragma unroll
        for (int r = 0; r < 8; ++r) {
            int row = row0 + widx * 16 + r + 8 * hi;
            int b_ = row >> 11;
            int l_ = row & (LL - 1);
            size_t bh = (size_t)b_ * HH + h;
            float v = c[t][r];
            if (sect == 0) {
                kbuf[(bh * LL + l_) * HD + d] = f2bf(v);
            } else if (sect == 1) {
                vtbuf[(bh * HD + d) * LL + l_] = f2bf(v);
            } else {
                qbuf[(bh * LL + l_) * HD + d] = f2bf(v * qscale);
            }
        }
    }
}

// ---------------------------------------------------------------------------
// K2: flash attention. One wave per (b, h, 16-row q tile); 4 waves / block.
// Register double-buffering: next K/V fragments load during current softmax.
// ---------------------------------------------------------------------------
union bfrag { v16bf v; uint4 q[2]; };
struct Tiles { bfrag k0l, k0h, k1l, k1h, vf[4]; };

__device__ __forceinline__ void load_bfrag(const unsigned short* p, bfrag& f) {
    f.q[0] = *(const uint4*)(p);
    f.q[1] = *(const uint4*)(p + 8);
}

__global__ __launch_bounds__(128) void attn_kernel(
        const unsigned short* __restrict__ qbuf,
        const unsigned short* __restrict__ kbuf,
        const unsigned short* __restrict__ vtbuf,
        const float* __restrict__ bias,
        float* __restrict__ out) {
    __shared__ unsigned short plds[4][16 * 32];   // P staging, 1KB per wave

    const int lane = threadIdx.x & 31;
    const int lo = lane & 15;
    const int hi = lane >> 4;
    const int widx = threadIdx.x >> 5;
    const int g = blockIdx.x * 4 + widx;          // 4096 wave tasks
    const int bh = g / (LL / 16);
    const int qt = g % (LL / 16);
    const int b_ = bh / HH;
    const int h  = bh % HH;
    const int qbase = qt * 16;

    const float* bias_h = bias + h * (2 * LL - 1);

    // Q A-fragments for both K=32 chunks of HD=64 (pre-scaled by 1/sqrt(HD))
    const unsigned short* qr = qbuf + ((size_t)bh * LL + qbase + lo) * HD;
    union { v16bf v; unsigned u[8]; } aq0, aq1;
#pragma unroll
    for (int p = 0; p < 8; ++p) {
        int d = 16 * hi + 2 * p;
        aq0.u[p] = *(const unsigned*)(qr + d);
        aq1.u[p] = *(const unsigned*)(qr + d + 32);
    }

    v8f o[4];
#pragma unroll
    for (int t = 0; t < 4; ++t) o[t] = (v8f){0.f,0.f,0.f,0.f,0.f,0.f,0.f,0.f};
    float mrow[8], lrow[8];
#pragma unroll
    for (int r = 0; r < 8; ++r) { mrow[r] = -INFINITY; lrow[r] = 0.0f; }

    const v8f zc = (v8f){0.f,0.f,0.f,0.f,0.f,0.f,0.f,0.f};

    Tiles cur, nxt;
    {   // preload jb = 0
        const unsigned short* kr0 = kbuf + ((size_t)bh * LL + lo) * HD + 16 * hi;
        load_bfrag(kr0, cur.k0l);
        load_bfrag(kr0 + 32, cur.k0h);
        load_bfrag(kr0 + 16 * HD, cur.k1l);
        load_bfrag(kr0 + 16 * HD + 32, cur.k1h);
#pragma unroll
        for (int t = 0; t < 4; ++t)
            load_bfrag(vtbuf + ((size_t)bh * HD + t * 16 + lo) * LL + 16 * hi, cur.vf[t]);
    }

    for (int jb = 0; jb < LL; jb += 32) {
        // ---- S = Q * K^T for 2 n-tiles (16 cols each) ----
        v8f s0 = __builtin_amdgcn_wmma_f32_16x16x32_bf16(
            false, aq0.v, false, cur.k0l.v, (short)0, zc, false, false);
        s0 = __builtin_amdgcn_wmma_f32_16x16x32_bf16(
            false, aq1.v, false, cur.k0h.v, (short)0, s0, false, false);
        v8f s1 = __builtin_amdgcn_wmma_f32_16x16x32_bf16(
            false, aq0.v, false, cur.k1l.v, (short)0, zc, false, false);
        s1 = __builtin_amdgcn_wmma_f32_16x16x32_bf16(
            false, aq1.v, false, cur.k1h.v, (short)0, s1, false, false);

        // ---- prefetch next iteration's K/V (overlaps softmax below) ----
        if (jb + 32 < LL) {
            const int jn = jb + 32;
            const unsigned short* kr0 = kbuf + ((size_t)bh * LL + jn + lo) * HD + 16 * hi;
            load_bfrag(kr0, nxt.k0l);
            load_bfrag(kr0 + 32, nxt.k0h);
            load_bfrag(kr0 + 16 * HD, nxt.k1l);
            load_bfrag(kr0 + 16 * HD + 32, nxt.k1h);
#pragma unroll
            for (int t = 0; t < 4; ++t)
                load_bfrag(vtbuf + ((size_t)bh * HD + t * 16 + lo) * LL + jn + 16 * hi,
                           nxt.vf[t]);
        }

        // ---- add bias ----
#pragma unroll
        for (int r = 0; r < 8; ++r) {
            int m = qbase + r + 8 * hi;
            s0[r] += bias_h[m - (jb + lo) + (LL - 1)];
            s1[r] += bias_h[m - (jb + 16 + lo) + (LL - 1)];
        }

        // ---- online softmax update ----
        float p0[8], p1[8];
#pragma unroll
        for (int r = 0; r < 8; ++r) {
            float mx = half16_max(fmaxf(s0[r], s1[r]));
            float mnew = fmaxf(mrow[r], mx);
            float alpha = __expf(mrow[r] - mnew);
            p0[r] = __expf(s0[r] - mnew);
            p1[r] = __expf(s1[r] - mnew);
            float rsum = half16_sum(p0[r] + p1[r]);
            lrow[r] = lrow[r] * alpha + rsum;
            mrow[r] = mnew;
#pragma unroll
            for (int t = 0; t < 4; ++t) o[t][r] *= alpha;
        }

        // ---- transpose P (C-layout) to A-layout via wave-private LDS ----
#pragma unroll
        for (int r = 0; r < 8; ++r) {
            int m = r + 8 * hi;
            plds[widx][m * 32 + lo]      = f2bf(p0[r]);
            plds[widx][m * 32 + 16 + lo] = f2bf(p1[r]);
        }
        union { v16bf v; unsigned u[8]; } pa;
#pragma unroll
        for (int p = 0; p < 8; ++p) {
            int kk = 16 * hi + 2 * p;
            pa.u[p] = *(const unsigned*)&plds[widx][lo * 32 + kk];
        }

        // ---- O += P * V ----
#pragma unroll
        for (int t = 0; t < 4; ++t) {
            o[t] = __builtin_amdgcn_wmma_f32_16x16x32_bf16(
                false, pa.v, false, cur.vf[t].v, (short)0, o[t], false, false);
        }

        cur = nxt;
    }

    // ---- normalize and write out[b, l, h*HD + d] (f32) ----
    float rinv[8];
#pragma unroll
    for (int r = 0; r < 8; ++r) rinv[r] = 1.0f / lrow[r];
#pragma unroll
    for (int t = 0; t < 4; ++t) {
#pragma unroll
        for (int r = 0; r < 8; ++r) {
            int m = qbase + r + 8 * hi;
            out[((size_t)b_ * LL + m) * DD + h * HD + t * 16 + lo] = o[t][r] * rinv[r];
        }
    }
}

// ---------------------------------------------------------------------------
extern "C" void kernel_launch(void* const* d_in, const int* in_sizes, int n_in,
                              void* d_out, int out_size, void* d_ws, size_t ws_size,
                              hipStream_t stream) {
    const float* x    = (const float*)d_in[0];
    const float* w_in = (const float*)d_in[1];
    const float* koff = (const float*)d_in[2];
    const float* kamp = (const float*)d_in[3];
    const float* kshp = (const float*)d_in[4];
    float* out = (float*)d_out;

    unsigned short* qbuf  = (unsigned short*)d_ws;
    unsigned short* kbuf  = qbuf + NQKV;
    unsigned short* vtbuf = kbuf + NQKV;
    float* bias = (float*)(vtbuf + NQKV);
    unsigned short* xb = (unsigned short*)(bias + NBIAS);
    unsigned short* wb = xb + NX;

    cvt_bf16_kernel<<<(NX / 4 + 255) / 256, 256, 0, stream>>>(x, xb, NX / 4);
    cvt_bf16_kernel<<<(NW / 4 + 255) / 256, 256, 0, stream>>>(w_in, wb, NW / 4);
    tisa_bias_kernel<<<(NBIAS + 255) / 256, 256, 0, stream>>>(koff, kamp, kshp, bias);
    // 32 M-blocks * 48 N-blocks, 8 waves per block
    proj_kernel<<<32 * 48, 256, 0, stream>>>(xb, wb, qbuf, kbuf, vtbuf);
    // 2*16*128 = 4096 wave tasks, 4 waves per block
    attn_kernel<<<4096 / 4, 128, 0, stream>>>(qbuf, kbuf, vtbuf, bias, out);
}
